// MultiHeadAttention_87393994539289
// MI455X (gfx1250) — compile-verified, matrix operand order checked by reading, not run on hardware
//
#include <hip/hip_runtime.h>
#include <hip/hip_bf16.h>

#define HD __device__ __forceinline__

// ---- model constants -------------------------------------------------------
#define BATCH   4
#define SEQ     1024
#define DMODEL  1024
#define NHEAD   16
#define DK      64
#define MROWS   (BATCH * SEQ)          // 4096
#define LN_EPSF 1e-5f

// ---- vector types ----------------------------------------------------------
typedef __attribute__((ext_vector_type(16))) __bf16          v16bf;
typedef __attribute__((ext_vector_type(8)))  float           v8f;
typedef __attribute__((ext_vector_type(4)))  float           v4f;
typedef __attribute__((ext_vector_type(8)))  unsigned short  v8us;
typedef __attribute__((ext_vector_type(4)))  unsigned short  v4us;

union ABFrag {
    v16bf v;
    v8us  h[2];
    unsigned short s[16];
};

// round-to-nearest-even fp32 -> bf16 (raw u16)
static HD unsigned short f2bf(float f) {
    unsigned int u = __float_as_uint(f);
    unsigned int r = (u + 0x7FFFu + ((u >> 16) & 1u)) >> 16;
    return (unsigned short)r;
}

// A/B fragment for v_wmma_f32_16x16x32_bf16 from K-major bf16 memory.
// Lane l<16 holds row (base_row+l), K chunks [k0, k0+7] and [k0+16, k0+23];
// lanes 16-31 hold the same rows shifted by K+8 (per ISA 16-bit A layout).
static HD ABFrag load_frag_bf16(const unsigned short* base, int row0, int lda, int k0) {
    int lane = threadIdx.x & 31;
    const unsigned short* p = base + (size_t)(row0 + (lane & 15)) * lda
                                   + k0 + ((lane >> 4) << 3);
    ABFrag f;
    f.h[0] = *(const v8us*)p;
    f.h[1] = *(const v8us*)(p + 16);
    return f;
}

// Same fragment but sourced from fp32 memory (converted in-register).
static HD ABFrag load_frag_f32(const float* base, int row0, int lda, int k0) {
    int lane = threadIdx.x & 31;
    const float* p = base + (size_t)(row0 + (lane & 15)) * lda
                          + k0 + ((lane >> 4) << 3);
    v4f x0 = *(const v4f*)(p);
    v4f x1 = *(const v4f*)(p + 4);
    v4f x2 = *(const v4f*)(p + 16);
    v4f x3 = *(const v4f*)(p + 20);
    ABFrag f;
    f.s[0] = f2bf(x0.x); f.s[1] = f2bf(x0.y); f.s[2]  = f2bf(x0.z); f.s[3]  = f2bf(x0.w);
    f.s[4] = f2bf(x1.x); f.s[5] = f2bf(x1.y); f.s[6]  = f2bf(x1.z); f.s[7]  = f2bf(x1.w);
    f.s[8] = f2bf(x2.x); f.s[9] = f2bf(x2.y); f.s[10] = f2bf(x2.z); f.s[11] = f2bf(x2.w);
    f.s[12] = f2bf(x3.x); f.s[13] = f2bf(x3.y); f.s[14] = f2bf(x3.z); f.s[15] = f2bf(x3.w);
    return f;
}

static HD v8f wmma_bf16(const ABFrag& a, const ABFrag& b, v8f c) {
    return __builtin_amdgcn_wmma_f32_16x16x32_bf16(
        /*neg_a=*/false, a.v, /*neg_b=*/false, b.v,
        /*c_mod=*/(short)0, c, /*reuse_a=*/false, /*reuse_b=*/false);
}

// ---------------------------------------------------------------------------
// fp32 -> bf16 elementwise conversion (n multiple of 4)
__global__ void k_f2bf(const float* __restrict__ in, unsigned short* __restrict__ out, int n) {
    int i = (blockIdx.x * blockDim.x + threadIdx.x) * 4;
    if (i < n) {
        v4f x = *(const v4f*)(in + i);
        v4us o;
        o.x = f2bf(x.x); o.y = f2bf(x.y); o.z = f2bf(x.z); o.w = f2bf(x.w);
        *(v4us*)(out + i) = o;
    }
}

// ---------------------------------------------------------------------------
// GEMM: out[M,N] = A[M,K] * W[N,K]^T + bias, bf16 output.
// mode 0: row-major [M,N] bf16. mode 1: transposed per-batch store for V:
//         vht[(b*SEQ + n)*SEQ + (m % SEQ)]  (key-contiguous rows).
// One wave per block; wave computes a 16x64 strip (A fragment reused 4x).
__global__ __launch_bounds__(32) void k_gemm_bf16out(
    const unsigned short* __restrict__ A, const unsigned short* __restrict__ W,
    const float* __restrict__ bias, unsigned short* __restrict__ out,
    int K, int N, int mode) {
    const int m0 = blockIdx.y * 16;
    const int n0 = blockIdx.x * 64;
    v8f c[4] = {};
    for (int k0 = 0; k0 < K; k0 += 32) {
        ABFrag a = load_frag_bf16(A, m0, K, k0);
#pragma unroll
        for (int j = 0; j < 4; ++j) {
            ABFrag b = load_frag_bf16(W, n0 + j * 16, K, k0);
            c[j] = wmma_bf16(a, b, c[j]);
        }
    }
    const int lane = threadIdx.x & 31;
    const int cn = lane & 15;
    const int rb = (lane >> 4) << 3;
    if (mode == 0) {
#pragma unroll
        for (int j = 0; j < 4; ++j) {
            int col = n0 + j * 16 + cn;
            float bcol = bias[col];
#pragma unroll
            for (int i = 0; i < 8; ++i)
                out[(size_t)(m0 + rb + i) * N + col] = f2bf(c[j][i] + bcol);
        }
    } else {
        const int bb = m0 >> 10;          // batch
        const int l0 = (m0 & 1023) + rb;  // seq pos of first row in this lane
#pragma unroll
        for (int j = 0; j < 4; ++j) {
            int col = n0 + j * 16 + cn;   // h*64 + d
            float bcol = bias[col];
            v8us pk;
#pragma unroll
            for (int i = 0; i < 8; ++i) pk[i] = f2bf(c[j][i] + bcol);
            *(v8us*)(out + ((size_t)(bb * SEQ + col)) * SEQ + l0) = pk;
        }
    }
}

// ---------------------------------------------------------------------------
// GEMM with fp32 output + bias (output projection, pre-residual).
__global__ __launch_bounds__(32) void k_gemm_f32out(
    const unsigned short* __restrict__ A, const unsigned short* __restrict__ W,
    const float* __restrict__ bias, float* __restrict__ out, int K, int N) {
    const int m0 = blockIdx.y * 16;
    const int n0 = blockIdx.x * 64;
    v8f c[4] = {};
    for (int k0 = 0; k0 < K; k0 += 32) {
        ABFrag a = load_frag_bf16(A, m0, K, k0);
#pragma unroll
        for (int j = 0; j < 4; ++j) {
            ABFrag b = load_frag_bf16(W, n0 + j * 16, K, k0);
            c[j] = wmma_bf16(a, b, c[j]);
        }
    }
    const int lane = threadIdx.x & 31;
    const int cn = lane & 15;
    const int rb = (lane >> 4) << 3;
#pragma unroll
    for (int j = 0; j < 4; ++j) {
        int col = n0 + j * 16 + cn;
        float bcol = bias[col];
#pragma unroll
        for (int i = 0; i < 8; ++i)
            out[(size_t)(m0 + rb + i) * N + col] = c[j][i] + bcol;
    }
}

// ---------------------------------------------------------------------------
// Fused scores + gate + mask + softmax.  grid = (SEQ/16, BATCH*NHEAD), 256 thr.
// Wave w computes keys [w*128, w*128+128) into a 16x1024 LDS strip,
// then the workgroup does a row softmax and writes attn once.
__global__ __launch_bounds__(256) void k_scores_softmax(
    const unsigned short* __restrict__ qh, const unsigned short* __restrict__ kh,
    const float* __restrict__ k_gate, const int* __restrict__ mask,
    float* __restrict__ attn) {
    __shared__ float sm[16 * 1028];
    __shared__ float red[16][17];
    __shared__ float stat[16];

    const int qt = blockIdx.x;
    const int bh = blockIdx.y;
    const int bb = bh >> 4;
    const int h  = bh & 15;
    const int q0 = qt * 16;
    const int wave = threadIdx.x >> 5;
    const int lane = threadIdx.x & 31;
    const int cn = lane & 15;
    const int rb = (lane >> 4) << 3;

    const unsigned short* Abase = qh + ((size_t)(bb * SEQ + q0)) * DMODEL + h * DK;
    const unsigned short* Bbase = kh + ((size_t)(bb * SEQ)) * DMODEL + h * DK;

    ABFrag a0 = load_frag_bf16(Abase, 0, DMODEL, 0);
    ABFrag a1 = load_frag_bf16(Abase, 0, DMODEL, 32);

    const float inv_temp = 0.125f;  // 1/sqrt(64)
    for (int t = 0; t < 8; ++t) {
        int key0 = wave * 128 + t * 16;
        ABFrag b0 = load_frag_bf16(Bbase, key0, DMODEL, 0);
        ABFrag b1 = load_frag_bf16(Bbase, key0, DMODEL, 32);
        v8f c = {};
        c = wmma_bf16(a0, b0, c);
        c = wmma_bf16(a1, b1, c);
        int key = key0 + cn;
#pragma unroll
        for (int i = 0; i < 8; ++i) {
            int qr = rb + i;
            float g = k_gate[((size_t)bh * SEQ + q0 + qr) * SEQ + key];
            int  mk = mask[((size_t)(bb * SEQ + q0 + qr)) * SEQ + key];
            sm[qr * 1028 + key] = mk ? -INFINITY : c[i] * inv_temp * g;
        }
    }
    __syncthreads();

    // row softmax: 16 threads per row, strided float4 (bank/coalesce friendly)
    const int r   = threadIdx.x >> 4;
    const int c16 = threadIdx.x & 15;
    float* rowp = sm + r * 1028;

    float mx = -INFINITY;
#pragma unroll
    for (int j = 0; j < 16; ++j) {
        v4f x = *(const v4f*)(rowp + c16 * 4 + 64 * j);
        mx = fmaxf(mx, fmaxf(fmaxf(x.x, x.y), fmaxf(x.z, x.w)));
    }
    red[r][c16] = mx;
    __syncthreads();
    if (c16 == 0) {
        float m2 = -INFINITY;
        for (int j = 0; j < 16; ++j) m2 = fmaxf(m2, red[r][j]);
        stat[r] = m2;
    }
    __syncthreads();
    mx = stat[r];
    __syncthreads();

    float sum = 0.f;
#pragma unroll
    for (int j = 0; j < 16; ++j) {
        float* p = rowp + c16 * 4 + 64 * j;
        v4f x = *(const v4f*)p;
        x.x = __expf(x.x - mx); x.y = __expf(x.y - mx);
        x.z = __expf(x.z - mx); x.w = __expf(x.w - mx);
        sum += x.x + x.y + x.z + x.w;
        *(v4f*)p = x;
    }
    red[r][c16] = sum;
    __syncthreads();
    if (c16 == 0) {
        float s2 = 0.f;
        for (int j = 0; j < 16; ++j) s2 += red[r][j];
        stat[r] = s2;
    }
    __syncthreads();
    float inv = 1.0f / stat[r];

    float* orow = attn + ((size_t)bh * SEQ + q0 + r) * SEQ;
#pragma unroll
    for (int j = 0; j < 16; ++j) {
        v4f x = *(const v4f*)(rowp + c16 * 4 + 64 * j);
        x.x *= inv; x.y *= inv; x.z *= inv; x.w *= inv;
        *(v4f*)(orow + c16 * 4 + 64 * j) = x;
    }
}

// ---------------------------------------------------------------------------
// attn @ V:  per (b,h): out[l, h*64+d] = sum_key attn[bh,l,key] * vht[(b,h,d),key]
// grid = (SEQ/16, BATCH*NHEAD), one wave; attn read from HBM exactly once.
__global__ __launch_bounds__(32) void k_av(
    const float* __restrict__ attn, const unsigned short* __restrict__ vht,
    unsigned short* __restrict__ av) {
    const int m0 = blockIdx.x * 16;  // query tile within batch
    const int bh = blockIdx.y;
    const int bb = bh >> 4;
    const int h  = bh & 15;
    const float* Abase = attn + ((size_t)bh * SEQ + m0) * SEQ;
    const unsigned short* Bbase = vht + ((size_t)(bb * SEQ + h * DK)) * SEQ;

    v8f c[4] = {};
    for (int k0 = 0; k0 < SEQ; k0 += 32) {
        ABFrag a = load_frag_f32(Abase, 0, SEQ, k0);
#pragma unroll
        for (int j = 0; j < 4; ++j) {
            ABFrag b = load_frag_bf16(Bbase, j * 16, SEQ, k0);
            c[j] = wmma_bf16(a, b, c[j]);
        }
    }
    const int lane = threadIdx.x & 31;
    const int cn = lane & 15;
    const int rb = (lane >> 4) << 3;
#pragma unroll
    for (int j = 0; j < 4; ++j) {
        int col = h * DK + j * 16 + cn;
#pragma unroll
        for (int i = 0; i < 8; ++i)
            av[((size_t)(bb * SEQ + m0 + rb + i)) * DMODEL + col] = f2bf(c[j][i]);
    }
}

// ---------------------------------------------------------------------------
// Residual + LayerNorm. One 256-thread block per row of 1024.
__global__ __launch_bounds__(256) void k_ln(
    const float* __restrict__ pre, const float* __restrict__ resid,
    const float* __restrict__ gamma, const float* __restrict__ beta,
    float* __restrict__ out) {
    __shared__ float s1[256];
    __shared__ float s2[256];
    const int row = blockIdx.x;
    const int t = threadIdx.x;
    v4f x = *(const v4f*)(pre + (size_t)row * DMODEL + t * 4);
    v4f rr = *(const v4f*)(resid + (size_t)row * DMODEL + t * 4);
    x.x += rr.x; x.y += rr.y; x.z += rr.z; x.w += rr.w;
    s1[t] = x.x + x.y + x.z + x.w;
    s2[t] = x.x * x.x + x.y * x.y + x.z * x.z + x.w * x.w;
    __syncthreads();
    for (int off = 128; off > 0; off >>= 1) {
        if (t < off) { s1[t] += s1[t + off]; s2[t] += s2[t + off]; }
        __syncthreads();
    }
    const float mean = s1[0] * (1.0f / DMODEL);
    const float var  = s2[0] * (1.0f / DMODEL) - mean * mean;
    const float rs = rsqrtf(var + LN_EPSF);
    v4f g = *(const v4f*)(gamma + t * 4);
    v4f bt = *(const v4f*)(beta + t * 4);
    v4f y;
    y.x = (x.x - mean) * rs * g.x + bt.x;
    y.y = (x.y - mean) * rs * g.y + bt.y;
    y.z = (x.z - mean) * rs * g.z + bt.z;
    y.w = (x.w - mean) * rs * g.w + bt.w;
    *(v4f*)(out + (size_t)row * DMODEL + t * 4) = y;
}

// ---------------------------------------------------------------------------
extern "C" void kernel_launch(void* const* d_in, const int* in_sizes, int n_in,
                              void* d_out, int out_size, void* d_ws, size_t ws_size,
                              hipStream_t stream) {
    const float* q      = (const float*)d_in[0];
    const float* k      = (const float*)d_in[1];
    const float* v      = (const float*)d_in[2];
    const float* k_gate = (const float*)d_in[3];
    const int*   mask   = (const int*)d_in[4];
    const float* Wq = (const float*)d_in[5];
    const float* bq = (const float*)d_in[6];
    const float* Wk = (const float*)d_in[7];
    const float* bk = (const float*)d_in[8];
    const float* Wv = (const float*)d_in[9];
    const float* bv = (const float*)d_in[10];
    const float* Wo = (const float*)d_in[11];
    const float* bo = (const float*)d_in[12];
    const float* gamma = (const float*)d_in[13];
    const float* beta  = (const float*)d_in[14];

    float* out  = (float*)d_out;
    float* attn = out + (size_t)MROWS * DMODEL;  // tuple part 2: [B,H,L,L]

    // workspace carve-up
    const size_t MK = (size_t)MROWS * DMODEL;        // 4096*1024
    const size_t WK = (size_t)DMODEL * DMODEL;       // 1024*1024
    char* ws = (char*)d_ws;
    auto carve = [&](size_t bytes) {
        char* p = ws;
        ws += (bytes + 255) & ~(size_t)255;
        return p;
    };
    unsigned short* qbf  = (unsigned short*)carve(MK * 2);
    unsigned short* kbf  = (unsigned short*)carve(MK * 2);
    unsigned short* vbf  = (unsigned short*)carve(MK * 2);
    unsigned short* wqbf = (unsigned short*)carve(WK * 2);
    unsigned short* wkbf = (unsigned short*)carve(WK * 2);
    unsigned short* wvbf = (unsigned short*)carve(WK * 2);
    unsigned short* wobf = (unsigned short*)carve(WK * 2);
    unsigned short* qh   = (unsigned short*)carve(MK * 2);
    unsigned short* kh   = (unsigned short*)carve(MK * 2);
    unsigned short* vht  = (unsigned short*)carve(MK * 2);
    unsigned short* av   = (unsigned short*)carve(MK * 2);
    float*          pre  = (float*)carve(MK * 4);

    // 1) bf16 conversions
    {
        int n = (int)MK;
        int blocks = (n / 4 + 255) / 256;
        k_f2bf<<<blocks, 256, 0, stream>>>(q, qbf, n);
        k_f2bf<<<blocks, 256, 0, stream>>>(k, kbf, n);
        k_f2bf<<<blocks, 256, 0, stream>>>(v, vbf, n);
        int nw = (int)WK;
        int wblocks = (nw / 4 + 255) / 256;
        k_f2bf<<<wblocks, 256, 0, stream>>>(Wq, wqbf, nw);
        k_f2bf<<<wblocks, 256, 0, stream>>>(Wk, wkbf, nw);
        k_f2bf<<<wblocks, 256, 0, stream>>>(Wv, wvbf, nw);
        k_f2bf<<<wblocks, 256, 0, stream>>>(Wo, wobf, nw);
    }

    // 2) Q/K/V projections (WMMA bf16, fp32 accumulate)
    {
        dim3 grid(DMODEL / 64, MROWS / 16);  // (16, 256)
        k_gemm_bf16out<<<grid, 32, 0, stream>>>(qbf, wqbf, bq, qh,  DMODEL, DMODEL, 0);
        k_gemm_bf16out<<<grid, 32, 0, stream>>>(kbf, wkbf, bk, kh,  DMODEL, DMODEL, 0);
        k_gemm_bf16out<<<grid, 32, 0, stream>>>(vbf, wvbf, bv, vht, DMODEL, DMODEL, 1);
    }

    // 3) scores + gate + mask + softmax -> attn (d_out)
    {
        dim3 grid(SEQ / 16, BATCH * NHEAD);  // (64, 64)
        k_scores_softmax<<<grid, 256, 0, stream>>>(qh, kh, k_gate, mask, attn);
    }

    // 4) attn @ V
    {
        dim3 grid(SEQ / 16, BATCH * NHEAD);  // (64, 64)
        k_av<<<grid, 32, 0, stream>>>(attn, vht, av);
    }

    // 5) output projection (+ bias)
    {
        dim3 grid(DMODEL / 64, MROWS / 16);
        k_gemm_f32out<<<grid, 32, 0, stream>>>(av, wobf, bo, pre, DMODEL, DMODEL);
    }

    // 6) residual + LayerNorm -> out (d_out)
    k_ln<<<MROWS, 256, 0, stream>>>(pre, q, gamma, beta, out);
}
